// TransformerBlock_17093969838599
// MI455X (gfx1250) — compile-verified, matrix operand order checked by reading, not run on hardware
//
#include <hip/hip_runtime.h>

// ---------------------------------------------------------------------------
// TransformerBlock forward for MI455X (gfx1250, wave32, WMMA + TDM).
// All matmuls run through v_wmma_f32_16x16x32_bf16 (bf16 in, f32 accum).
// Tile staging to LDS uses the Tensor Data Mover (tensor_load_to_lds),
// double-buffered in the GEMMs and synchronized with s_wait_tensorcnt.
// Pipeline (all on `stream`, scratch in d_ws):
//   1) cast weights fp32->bf16
//   2) h  = rmsnorm(x, ln1_w)            -> bf16
//   3) q,k,v = h @ W{q,k,v}^T            (WMMA GEMM, bf16 out)
//   4) attn = flash-attention (causal)    (WMMA QK^T and PV, online softmax)
//   5) x1 = x + attn @ Wo^T               (WMMA GEMM, f32 out + residual)
//   6) h2 = rmsnorm(x1, ln2_w)           -> bf16
//   7) ff = gelu_exact(h2 @ Wup^T)        (WMMA GEMM, erf GELU, bf16 out)
//   8) out = x1 + ff @ Wdown^T            (WMMA GEMM, f32 out + residual)
// ---------------------------------------------------------------------------

typedef __bf16 bf16;
typedef __attribute__((ext_vector_type(16))) __bf16 bf16x16;
typedef __attribute__((ext_vector_type(8)))  float   v8f;
typedef __attribute__((ext_vector_type(4)))  unsigned int u32x4;
typedef __attribute__((ext_vector_type(8)))  int     i32x8;
typedef __attribute__((ext_vector_type(4)))  int     i32x4;

constexpr int B_   = 2;
constexpr int L_   = 2048;
constexpr int D_   = 2048;
constexpr int H_   = 16;
constexpr int DH_  = 128;
constexpr int DFF_ = 8192;
constexpr int M_   = B_ * L_;   // 4096 rows

__device__ __forceinline__ v8f vzero() {
    v8f z = {0.f, 0.f, 0.f, 0.f, 0.f, 0.f, 0.f, 0.f};
    return z;
}

__device__ __forceinline__ v8f wmma_bf16(bf16x16 a, bf16x16 b, v8f c) {
    // 8 args: (neg_a, A, neg_b, B, c_mod, C, reuse_a, reuse_b)
    return __builtin_amdgcn_wmma_f32_16x16x32_bf16(false, a, false, b, (short)0, c,
                                                   false, false);
}

// LDS aperture addresses carry the LDS byte offset in the low 32 bits.
__device__ __forceinline__ unsigned lds_addr_of(const void* p) {
    return (unsigned)(unsigned long long)p;
}

// ---------------------------------------------------------------------------
// TDM: DMA a 2D tile (rows x elems_x of bf16) from global memory to LDS.
// Descriptor packing per CDNA5 ISA (D# groups 0/1):
//   g0: count=1 | lds_addr | global_addr[56:0] | type=2
//   g1: data_size=2B, tensor_dim0=elems_x, tensor_dim1=rows,
//       tile_dim0=elems_x, tile_dim1=rows, tensor_dim0_stride=row_stride
// Tracked by TENSORcnt; pair with __builtin_amdgcn_s_wait_tensorcnt.
// ---------------------------------------------------------------------------
__device__ __forceinline__ void tdm_load_2d_bf16(unsigned lds_byte_addr,
                                                 const void* gptr,
                                                 unsigned elems_x, unsigned rows,
                                                 unsigned row_stride_elems) {
    unsigned long long ga = (unsigned long long)gptr;
    u32x4 g0;
    g0[0] = 1u;                                            // count=1, user-mode
    g0[1] = lds_byte_addr;                                 // lds_addr
    g0[2] = (unsigned)(ga & 0xffffffffu);                  // global_addr lo
    g0[3] = (unsigned)((ga >> 32) & 0x01ffffffu) | 0x80000000u;  // hi | type=2

    i32x8 g1;
    g1[0] = (int)(1u << 16);                               // data_size=1 (2 bytes)
    g1[1] = (int)((elems_x & 0xffffu) << 16);              // tensor_dim0 lo16
    g1[2] = (int)((elems_x >> 16) & 0xffffu) |
            (int)((rows & 0xffffu) << 16);                 // dim0 hi | dim1 lo
    g1[3] = (int)((rows >> 16) & 0xffffu) |
            (int)((elems_x & 0xffffu) << 16);              // dim1 hi | tile_dim0
    g1[4] = (int)(rows & 0xffffu);                         // tile_dim1 (tile_dim2=0)
    g1[5] = (int)row_stride_elems;                         // tensor_dim0_stride lo32
    g1[6] = 0;                                             // stride hi16, dim1_stride
    g1[7] = 0;

    i32x4 z4 = {0, 0, 0, 0};
#if __clang_major__ >= 23
    i32x8 z8 = {0, 0, 0, 0, 0, 0, 0, 0};
    __builtin_amdgcn_tensor_load_to_lds(g0, g1, z4, z4, z8, 0);
#else
    __builtin_amdgcn_tensor_load_to_lds(g0, g1, z4, z4, 0);
#endif
}

// ---------------------------------------------------------------------------
// fp32 -> bf16 cast (weights), 4 elements / thread
// ---------------------------------------------------------------------------
__global__ __launch_bounds__(256)
void cast_f32_bf16(const float* __restrict__ s, bf16* __restrict__ d, int n4) {
    int i = blockIdx.x * blockDim.x + threadIdx.x;
    if (i < n4) {
        __builtin_prefetch(s + 4 * (size_t)i + 4096, 0, 0);  // global_prefetch_b8
        float4 v = reinterpret_cast<const float4*>(s)[i];
        d[4 * i + 0] = (bf16)v.x;
        d[4 * i + 1] = (bf16)v.y;
        d[4 * i + 2] = (bf16)v.z;
        d[4 * i + 3] = (bf16)v.w;
    }
}

// ---------------------------------------------------------------------------
// RMSNorm row kernel: out_bf16 = x / rms(x) * w   (one block per row, D_=2048)
// ---------------------------------------------------------------------------
__global__ __launch_bounds__(256)
void rmsnorm_cast(const float* __restrict__ x, const float* __restrict__ w,
                  bf16* __restrict__ out) {
    __shared__ float red[256];
    const int row = blockIdx.x;
    const float* xr = x + (size_t)row * D_;
    float ss = 0.f;
    for (int i = threadIdx.x; i < D_; i += 256) {
        float v = xr[i];
        ss += v * v;
    }
    red[threadIdx.x] = ss;
    __syncthreads();
    for (int s = 128; s > 0; s >>= 1) {
        if (threadIdx.x < s) red[threadIdx.x] += red[threadIdx.x + s];
        __syncthreads();
    }
    const float inv = 1.f / sqrtf(red[0] / (float)D_ + 1e-8f);
    bf16* orow = out + (size_t)row * D_;
    for (int i = threadIdx.x; i < D_; i += 256) {
        orow[i] = (bf16)(xr[i] * inv * w[i]);
    }
}

// ---------------------------------------------------------------------------
// WMMA GEMM: C[M,N] = A[M,K] * W[N,K]^T    (torch Linear layout)
// Block tile 128x64, 8 waves, each wave a 32x32 tile (2x2 of 16x16 WMMA).
// TDM double-buffered LDS staging: wave 0 issues the next K-slab's pair of
// tensor_load_to_lds, all waves s_wait_tensorcnt + barrier, then WMMA.
// EPI: 0 = store bf16, 1 = residual(f32) + store f32, 2 = erf-GELU + store bf16
// ---------------------------------------------------------------------------
template <int EPI>
__global__ __launch_bounds__(256)
void gemm_bf16_wmma(const bf16* __restrict__ A, const bf16* __restrict__ Bw,
                    const float* __restrict__ Res, float* __restrict__ outF,
                    bf16* __restrict__ outB, int Ndim, int Kdim) {
    __shared__ __attribute__((aligned(128))) bf16 As[2][128 * 32];
    __shared__ __attribute__((aligned(128))) bf16 Bs[2][64 * 32];

    const int m0   = blockIdx.y * 128;
    const int n0   = blockIdx.x * 64;
    const int t    = threadIdx.x;
    const int lane = t & 31;
    const int wave = t >> 5;
    const int wm   = (wave & 3) * 32;   // 4 waves along M
    const int wn   = (wave >> 2) * 32;  // 2 waves along N
    const int fr   = lane & 15;         // fragment row within 16
    const int fo   = (lane >> 4) * 16;  // fragment K offset (0 or 16)
    const bool ldw = (wave == 0);       // TDM issuing wave (EXEC-independent op)

    v8f acc[2][2];
    for (int i = 0; i < 2; ++i)
        for (int j = 0; j < 2; ++j) acc[i][j] = vzero();

    // prologue: DMA K-slab 0 into buffer 0
    if (ldw) {
        tdm_load_2d_bf16(lds_addr_of(&As[0][0]), &A[(size_t)m0 * Kdim], 32, 128, Kdim);
        tdm_load_2d_bf16(lds_addr_of(&Bs[0][0]), &Bw[(size_t)n0 * Kdim], 32, 64, Kdim);
    }

    for (int kk = 0, it = 0; kk < Kdim; kk += 32, ++it) {
        const int  bi   = it & 1;
        const bool more = (kk + 32) < Kdim;
        if (ldw && more) {
            tdm_load_2d_bf16(lds_addr_of(&As[bi ^ 1][0]),
                             &A[(size_t)m0 * Kdim + kk + 32], 32, 128, Kdim);
            tdm_load_2d_bf16(lds_addr_of(&Bs[bi ^ 1][0]),
                             &Bw[(size_t)n0 * Kdim + kk + 32], 32, 64, Kdim);
        }
        if (more) __builtin_amdgcn_s_wait_tensorcnt(2);   // current pair done
        else      __builtin_amdgcn_s_wait_tensorcnt(0);   // drain on last slab
        __syncthreads();

        bf16x16 a0 = *(const bf16x16*)&As[bi][(wm + fr) * 32 + fo];
        bf16x16 a1 = *(const bf16x16*)&As[bi][(wm + 16 + fr) * 32 + fo];
        bf16x16 b0 = *(const bf16x16*)&Bs[bi][(wn + fr) * 32 + fo];
        bf16x16 b1 = *(const bf16x16*)&Bs[bi][(wn + 16 + fr) * 32 + fo];

        acc[0][0] = wmma_bf16(a0, b0, acc[0][0]);
        acc[0][1] = wmma_bf16(a0, b1, acc[0][1]);
        acc[1][0] = wmma_bf16(a1, b0, acc[1][0]);
        acc[1][1] = wmma_bf16(a1, b1, acc[1][1]);
        __syncthreads();   // buffer bi free for the DMA issued next iter
    }

    // Epilogue. C layout: VGPR r, lanes 0-15 -> M=r, lanes 16-31 -> M=r+8.
    for (int i = 0; i < 2; ++i)
        for (int j = 0; j < 2; ++j)
            for (int r = 0; r < 8; ++r) {
                int gr = m0 + wm + i * 16 + r + 8 * (lane >> 4);
                int gc = n0 + wn + j * 16 + (lane & 15);
                size_t idx = (size_t)gr * Ndim + gc;
                float v = acc[i][j][r];
                if constexpr (EPI == 0) {
                    outB[idx] = (bf16)v;
                } else if constexpr (EPI == 1) {
                    outF[idx] = Res[idx] + v;
                } else {
                    float g = 0.5f * v * (1.f + erff(v * 0.70710678118654752f));
                    outB[idx] = (bf16)g;
                }
            }
}

// ---------------------------------------------------------------------------
// Flash attention, causal. Grid: B_*H_*(L_/64) blocks, 128 threads (4 waves).
// K and V tiles arrive by TDM; V transposed once in LDS for the PV B-operand.
// Each wave owns 16 q rows; Q*K^T and P*V via WMMA; online softmax in regs.
// ---------------------------------------------------------------------------
__global__ __launch_bounds__(128)
void attn_fwd(const bf16* __restrict__ qg, const bf16* __restrict__ kg,
              const bf16* __restrict__ vg, bf16* __restrict__ og) {
    __shared__ __attribute__((aligned(128))) bf16 Ks[64 * DH_];   // [kpos][d]
    __shared__ __attribute__((aligned(128))) bf16 Vs[64 * DH_];   // [kpos][d]
    __shared__ __attribute__((aligned(128))) bf16 Vt[DH_ * 64];   // [d][kpos]
    __shared__ __attribute__((aligned(128))) bf16 Ps[4][16 * 64]; // per-wave P

    constexpr int NQT = L_ / 64;
    const int bid = blockIdx.x;
    const int qt  = bid % NQT;
    const int h   = (bid / NQT) % H_;
    const int b   = bid / (NQT * H_);
    const int q0  = qt * 64;

    const int t    = threadIdx.x;
    const int lane = t & 31;
    const int w    = t >> 5;            // wave 0..3
    const int fr   = lane & 15;
    const int fh   = lane >> 4;         // 0 or 1
    const bool ldw = (w == 0);

    // Q fragments: rows q0 + w*16 + fr, K-dim (=DH_) chunks of 32
    bf16x16 qf[4];
    const size_t qrowA = (size_t)(b * L_ + q0 + w * 16 + fr);
    for (int c = 0; c < 4; ++c)
        qf[c] = *(const bf16x16*)&qg[qrowA * D_ + h * DH_ + c * 32 + fh * 16];

    v8f o[8];
    for (int dt = 0; dt < 8; ++dt) o[dt] = vzero();
    float mrow[8], lrow[8];
    for (int r = 0; r < 8; ++r) { mrow[r] = -1e30f; lrow[r] = 0.f; }

    const float sc = 0.08838834764831845f;  // 1/sqrt(128)

    for (int kt = 0; kt <= qt; ++kt) {
        __syncthreads();  // everyone done with Ks/Vs/Vt/Ps of previous tile
        if (ldw) {
            const bf16* kbase = &kg[(size_t)(b * L_ + kt * 64) * D_ + h * DH_];
            const bf16* vbase = &vg[(size_t)(b * L_ + kt * 64) * D_ + h * DH_];
            tdm_load_2d_bf16(lds_addr_of(&Ks[0]), kbase, DH_, 64, D_);
            tdm_load_2d_bf16(lds_addr_of(&Vs[0]), vbase, DH_, 64, D_);
        }
        __builtin_amdgcn_s_wait_tensorcnt(0);
        __syncthreads();

        // transpose V tile in LDS: Vt[d][kpos] = Vs[kpos][d]
        for (int i = 0; i < 64; ++i) {
            int e = t + i * 128;
            int kp = e >> 7, d = e & 127;
            Vt[d * 64 + kp] = Vs[kp * DH_ + d];
        }
        __syncthreads();

        // S = Q*K^T : 4 tiles of 16 kpos, K-dim 128 in 4 WMMA steps
        v8f s[4];
        for (int n = 0; n < 4; ++n) {
            v8f a = vzero();
            for (int c = 0; c < 4; ++c) {
                bf16x16 bk = *(const bf16x16*)&Ks[(n * 16 + fr) * DH_ +
                                                  c * 32 + fh * 16];
                a = wmma_bf16(qf[c], bk, a);
            }
            s[n] = a;
        }

        // scale + causal mask. element (r): M = r + 8*fh, N = fr
        for (int n = 0; n < 4; ++n)
            for (int r = 0; r < 8; ++r) {
                float sv = s[n][r] * sc;
                int kpos = kt * 64 + n * 16 + fr;
                int qpos = q0 + w * 16 + r + 8 * fh;
                if (kpos > qpos) sv = -1e30f;
                s[n][r] = sv;
            }

        // online softmax per row; row spans 16 lanes of one half
        for (int r = 0; r < 8; ++r) {
            float tm = s[0][r];
            for (int n = 1; n < 4; ++n) tm = fmaxf(tm, s[n][r]);
            for (int off = 8; off > 0; off >>= 1)
                tm = fmaxf(tm, __shfl_xor(tm, off, 16));
            float mnew = fmaxf(mrow[r], tm);
            float corr = expf(mrow[r] - mnew);
            float psum = 0.f;
            for (int n = 0; n < 4; ++n) {
                float p = expf(s[n][r] - mnew);
                s[n][r] = p;
                psum += p;
            }
            for (int off = 8; off > 0; off >>= 1)
                psum += __shfl_xor(psum, off, 16);
            lrow[r] = lrow[r] * corr + psum;
            mrow[r] = mnew;
            for (int dt = 0; dt < 8; ++dt) o[dt][r] *= corr;
        }

        // P -> LDS (bf16), per-wave region, then reload in A-operand layout
        for (int r = 0; r < 8; ++r)
            for (int n = 0; n < 4; ++n)
                Ps[w][(r + 8 * fh) * 64 + n * 16 + fr] = (bf16)s[n][r];
        __syncthreads();

        // O += P * V : N = DH_ (8 tiles of 16), K-dim 64 in 2 WMMA steps
        for (int pc = 0; pc < 2; ++pc) {
            bf16x16 pa = *(const bf16x16*)&Ps[w][fr * 64 + pc * 32 + fh * 16];
            for (int dt = 0; dt < 8; ++dt) {
                bf16x16 vb = *(const bf16x16*)&Vt[(dt * 16 + fr) * 64 +
                                                  pc * 32 + fh * 16];
                o[dt] = wmma_bf16(pa, vb, o[dt]);
            }
        }
    }

    float inv[8];
    for (int r = 0; r < 8; ++r) inv[r] = 1.f / lrow[r];
    for (int dt = 0; dt < 8; ++dt)
        for (int r = 0; r < 8; ++r) {
            size_t row = (size_t)(b * L_ + q0 + w * 16 + r + 8 * fh);
            og[row * D_ + h * DH_ + dt * 16 + fr] = (bf16)(o[dt][r] * inv[r]);
        }
}

// ---------------------------------------------------------------------------
// Host launcher
// ---------------------------------------------------------------------------
extern "C" void kernel_launch(void* const* d_in, const int* in_sizes, int n_in,
                              void* d_out, int out_size, void* d_ws, size_t ws_size,
                              hipStream_t stream) {
    (void)in_sizes; (void)n_in; (void)out_size; (void)ws_size;

    const float* x    = (const float*)d_in[0];
    // d_in[1] = additive mask: causal mask is applied analytically in attn_fwd
    const float* ln1  = (const float*)d_in[2];
    const float* wq   = (const float*)d_in[3];
    const float* wk   = (const float*)d_in[4];
    const float* wv   = (const float*)d_in[5];
    const float* wo   = (const float*)d_in[6];
    const float* ln2  = (const float*)d_in[7];
    const float* wup  = (const float*)d_in[8];
    const float* wdn  = (const float*)d_in[9];
    float* out = (float*)d_out;

    // workspace carve-up (~288 MiB total)
    char* ws = (char*)d_ws;
    size_t off = 0;
    auto carve = [&](size_t bytes) {
        void* p = ws + off;
        off = (off + bytes + 255) & ~(size_t)255;
        return p;
    };
    bf16* wq_b  = (bf16*)carve((size_t)D_ * D_ * 2);
    bf16* wk_b  = (bf16*)carve((size_t)D_ * D_ * 2);
    bf16* wv_b  = (bf16*)carve((size_t)D_ * D_ * 2);
    bf16* wo_b  = (bf16*)carve((size_t)D_ * D_ * 2);
    bf16* wup_b = (bf16*)carve((size_t)DFF_ * D_ * 2);
    bf16* wdn_b = (bf16*)carve((size_t)DFF_ * D_ * 2);
    bf16* h_b   = (bf16*)carve((size_t)M_ * D_ * 2);
    bf16* q_b   = (bf16*)carve((size_t)M_ * D_ * 2);
    bf16* k_b   = (bf16*)carve((size_t)M_ * D_ * 2);
    bf16* v_b   = (bf16*)carve((size_t)M_ * D_ * 2);
    bf16* a_b   = (bf16*)carve((size_t)M_ * D_ * 2);
    float* x1   = (float*)carve((size_t)M_ * D_ * 4);
    bf16* h2_b  = (bf16*)carve((size_t)M_ * D_ * 2);
    bf16* ff_b  = (bf16*)carve((size_t)M_ * DFF_ * 2);

    // 1) weights -> bf16
    {
        int n4s = (D_ * D_) / 4;         // 1,048,576
        int n4l = (DFF_ * D_) / 4;       // 4,194,304
        dim3 blk(256);
        cast_f32_bf16<<<dim3((n4s + 255) / 256), blk, 0, stream>>>(wq,  wq_b,  n4s);
        cast_f32_bf16<<<dim3((n4s + 255) / 256), blk, 0, stream>>>(wk,  wk_b,  n4s);
        cast_f32_bf16<<<dim3((n4s + 255) / 256), blk, 0, stream>>>(wv,  wv_b,  n4s);
        cast_f32_bf16<<<dim3((n4s + 255) / 256), blk, 0, stream>>>(wo,  wo_b,  n4s);
        cast_f32_bf16<<<dim3((n4l + 255) / 256), blk, 0, stream>>>(wup, wup_b, n4l);
        cast_f32_bf16<<<dim3((n4l + 255) / 256), blk, 0, stream>>>(wdn, wdn_b, n4l);
    }

    // 2) h = rmsnorm(x, ln1)
    rmsnorm_cast<<<dim3(M_), dim3(256), 0, stream>>>(x, ln1, h_b);

    // 3) q,k,v projections
    {
        dim3 grid(D_ / 64, M_ / 128), blk(256);
        gemm_bf16_wmma<0><<<grid, blk, 0, stream>>>(h_b, wq_b, nullptr, nullptr, q_b, D_, D_);
        gemm_bf16_wmma<0><<<grid, blk, 0, stream>>>(h_b, wk_b, nullptr, nullptr, k_b, D_, D_);
        gemm_bf16_wmma<0><<<grid, blk, 0, stream>>>(h_b, wv_b, nullptr, nullptr, v_b, D_, D_);
    }

    // 4) attention
    attn_fwd<<<dim3(B_ * H_ * (L_ / 64)), dim3(128), 0, stream>>>(q_b, k_b, v_b, a_b);

    // 5) x1 = x + attn @ Wo^T
    gemm_bf16_wmma<1><<<dim3(D_ / 64, M_ / 128), dim3(256), 0, stream>>>(
        a_b, wo_b, x, x1, nullptr, D_, D_);

    // 6) h2 = rmsnorm(x1, ln2)
    rmsnorm_cast<<<dim3(M_), dim3(256), 0, stream>>>(x1, ln2, h2_b);

    // 7) ff = gelu(h2 @ Wup^T)
    gemm_bf16_wmma<2><<<dim3(DFF_ / 64, M_ / 128), dim3(256), 0, stream>>>(
        h2_b, wup_b, nullptr, nullptr, ff_b, DFF_, D_);

    // 8) out = x1 + ff @ Wdown^T
    gemm_bf16_wmma<1><<<dim3(D_ / 64, M_ / 128), dim3(256), 0, stream>>>(
        ff_b, wdn_b, x1, out, nullptr, D_, DFF_);
}
